// FMoE_69733089018080
// MI455X (gfx1250) — matrix operand order repaired and checked by reading, not compiled, not run on hardware
//
#include <hip/hip_runtime.h>
#include <hip/hip_bf16.h>
#include <math.h>

// Problem constants (match reference)
#define T_TOK   16384
#define D_MODEL 512
#define H_DIM   1024
#define N_EXP   8
#define TOP_K   2
#define N_ROWS  (T_TOK * TOP_K)   // 32768 dispatched rows

// GEMM tiling
#define BM 64
#define BN 64
#define BK 32
#define GTHREADS 128

typedef __attribute__((ext_vector_type(16))) __bf16 v16bf;
typedef __attribute__((ext_vector_type(8)))  __bf16 v8bf;
typedef __attribute__((ext_vector_type(4)))  __bf16 v4bf;
typedef __attribute__((ext_vector_type(2)))  __bf16 v2bf;
typedef __attribute__((ext_vector_type(8)))  float  v8f;

union Frag {
  v16bf v;
  v8bf  h[2];
};

// ---------------------------------------------------------------------------
// 0) zero per-expert counters
// ---------------------------------------------------------------------------
__global__ void init_kernel(int* __restrict__ cnt, int* __restrict__ fill) {
  if (threadIdx.x < N_EXP) {
    cnt[threadIdx.x]  = 0;
    fill[threadIdx.x] = 0;
  }
}

// ---------------------------------------------------------------------------
// 1) gate: logits = x @ gate_w + gate_b ; top-2 ; softmax over the 2 logits
//    one wave32 per token; gate_w staged in LDS (16 KB, reused by 8 waves)
// ---------------------------------------------------------------------------
__global__ __launch_bounds__(256)
void gate_kernel(const float* __restrict__ x,
                 const float* __restrict__ gw,   // [D_MODEL, N_EXP]
                 const float* __restrict__ gb,   // [N_EXP]
                 int*   __restrict__ top_idx,    // [N_ROWS]
                 float* __restrict__ scores,     // [N_ROWS]
                 int*   __restrict__ cnt)        // [N_EXP]
{
  __shared__ float sgw[D_MODEL * N_EXP];
  for (int i = threadIdx.x; i < D_MODEL * N_EXP; i += 256) sgw[i] = gw[i];
  __syncthreads();

  const int wid  = threadIdx.x >> 5;
  const int lane = threadIdx.x & 31;
  const int t    = blockIdx.x * 8 + wid;

  float acc[N_EXP];
#pragma unroll
  for (int e = 0; e < N_EXP; ++e) acc[e] = 0.f;

  const float* xr = x + (size_t)t * D_MODEL;
  for (int d = lane; d < D_MODEL; d += 32) {
    float xv = xr[d];
    const float* g = &sgw[d * N_EXP];
#pragma unroll
    for (int e = 0; e < N_EXP; ++e) acc[e] += xv * g[e];
  }
  // butterfly reduce across wave32
#pragma unroll
  for (int e = 0; e < N_EXP; ++e) {
#pragma unroll
    for (int off = 16; off > 0; off >>= 1)
      acc[e] += __shfl_xor(acc[e], off, 32);
  }

  if (lane == 0) {
    float logit[N_EXP];
#pragma unroll
    for (int e = 0; e < N_EXP; ++e) logit[e] = acc[e] + gb[e];

    int   i0 = 0;       float v0 = logit[0];
#pragma unroll
    for (int e = 1; e < N_EXP; ++e)
      if (logit[e] > v0) { v0 = logit[e]; i0 = e; }   // strict > => lower idx on ties
    int   i1 = -1;      float v1 = -3.4e38f;
#pragma unroll
    for (int e = 0; e < N_EXP; ++e)
      if (e != i0 && logit[e] > v1) { v1 = logit[e]; i1 = e; }

    float p1 = __expf(v1 - v0);       // v0 >= v1, numerically safe
    float s0 = 1.f / (1.f + p1);
    float s1 = p1 * s0;

    top_idx[t * 2 + 0] = i0;  top_idx[t * 2 + 1] = i1;
    scores [t * 2 + 0] = s0;  scores [t * 2 + 1] = s1;
    atomicAdd(&cnt[i0], 1);
    atomicAdd(&cnt[i1], 1);
  }
}

// ---------------------------------------------------------------------------
// 2) exclusive-scan of expert counts (E = 8, single thread)
// ---------------------------------------------------------------------------
__global__ void offsets_kernel(const int* __restrict__ cnt, int* __restrict__ offs) {
  int run = 0;
  for (int e = 0; e < N_EXP; ++e) { offs[e] = run; run += cnt[e]; }
}

// ---------------------------------------------------------------------------
// 3) scatter dispatched rows into expert-sorted xs; record pos per (t, slot)
// ---------------------------------------------------------------------------
__global__ __launch_bounds__(128)
void scatter_kernel(const float* __restrict__ x,
                    const int*   __restrict__ top_idx,
                    const int*   __restrict__ offs,
                    int*         __restrict__ fill,
                    int*         __restrict__ pos,
                    float*       __restrict__ xs)
{
  const int r = blockIdx.x;          // dispatched row id = t*K + k
  const int t = r >> 1;
  __shared__ int sp;
  if (threadIdx.x == 0) {
    int e = top_idx[r];
    int p = offs[e] + atomicAdd(&fill[e], 1);
    pos[r] = p;
    sp = p;
  }
  __syncthreads();
  const int p = sp;
  const float4* src = (const float4*)(x  + (size_t)t * D_MODEL);
  float4*       dst = (float4*)      (xs + (size_t)p * D_MODEL);
  dst[threadIdx.x] = src[threadIdx.x];          // 128 threads x float4 = 512 f
}

// ---------------------------------------------------------------------------
// 4/5) grouped GEMM via bf16x3 split-precision WMMA, + bias (+ exact GELU)
//      block = 128 threads (4 waves), tile 64(M) x 64(N), BK = 32
//      LDS double-buffered; packed fp32 -> bf16(hi/lo) staging
// ---------------------------------------------------------------------------
template <bool APPLY_GELU>
__global__ __launch_bounds__(GTHREADS)
void moe_grouped_gemm(const float* __restrict__ Xs,     // [N_ROWS, Din] expert-sorted
                      const float* __restrict__ W,      // [E, Din, Dout]
                      const float* __restrict__ Bias,   // [E, Dout]
                      float*       __restrict__ Out,    // [N_ROWS, Dout]
                      const int*   __restrict__ offs,
                      const int*   __restrict__ cnt,
                      int Din, int Dout)
{
  const int e     = blockIdx.z;
  const int cntE  = cnt[e];
  const int mBase = blockIdx.x * BM;
  if (mBase >= cntE) return;
  const int offE  = offs[e];
  const int n0    = blockIdx.y * BN;
  const float* Wp = W + (size_t)e * Din * Dout;

  // bf16 hi/lo staging; stride 40 elems = 80 B (multiple of 16 -> aligned b128)
  __shared__ __align__(16) __bf16 Ahi[2][BM][40];
  __shared__ __align__(16) __bf16 Alo[2][BM][40];
  __shared__ __align__(16) __bf16 Bhi[2][BN][40];   // [n][k] (transposed)
  __shared__ __align__(16) __bf16 Blo[2][BN][40];

  const int tid  = threadIdx.x;
  const int wid  = tid >> 5;
  const int lane = tid & 31;
  const int l15  = lane & 15;
  const int up   = (lane >= 16);

  // ---- prefetch registers (next K-tile) ------------------------------------
  float4 av[4];            // A: 4 quads/thread, quad = (row i, k-quad kq)
  float4 bv0[2], bv1[2];   // B: 2 blocks/thread, block = 2(k) x 4(n)

  auto prefetch = [&](int k0) {
#pragma unroll
    for (int j = 0; j < 4; ++j) {
      int idx = tid + j * GTHREADS;          // [0, 512)
      int i = idx >> 3, kq = idx & 7;
      int m = mBase + i;
      av[j] = (m < cntE)
                ? *(const float4*)&Xs[(size_t)(offE + m) * Din + k0 + kq * 4]
                : make_float4(0.f, 0.f, 0.f, 0.f);
    }
#pragma unroll
    for (int j = 0; j < 2; ++j) {
      int idx = tid + j * GTHREADS;          // [0, 256)
      int kp = idx >> 4, nq = idx & 15;
      const float* p = &Wp[(size_t)(k0 + kp * 2) * Dout + n0 + nq * 4];
      bv0[j] = *(const float4*)p;
      bv1[j] = *(const float4*)(p + Dout);
    }
  };

  auto stage = [&](int buf) {
#pragma unroll
    for (int j = 0; j < 4; ++j) {
      int idx = tid + j * GTHREADS;
      int i = idx >> 3, kq = idx & 7;
      float f[4] = { av[j].x, av[j].y, av[j].z, av[j].w };
      v4bf hi, lo;
#pragma unroll
      for (int u = 0; u < 4; ++u) {
        hi[u] = (__bf16)f[u];
        lo[u] = (__bf16)(f[u] - (float)hi[u]);
      }
      *(v4bf*)&Ahi[buf][i][kq * 4] = hi;     // ds_store_b64
      *(v4bf*)&Alo[buf][i][kq * 4] = lo;
    }
#pragma unroll
    for (int j = 0; j < 2; ++j) {
      int idx = tid + j * GTHREADS;
      int kp = idx >> 4, nq = idx & 15;
      float f0[4] = { bv0[j].x, bv0[j].y, bv0[j].z, bv0[j].w };
      float f1[4] = { bv1[j].x, bv1[j].y, bv1[j].z, bv1[j].w };
#pragma unroll
      for (int u = 0; u < 4; ++u) {
        int n = nq * 4 + u;
        v2bf hi, lo;
        hi[0] = (__bf16)f0[u];
        hi[1] = (__bf16)f1[u];
        lo[0] = (__bf16)(f0[u] - (float)hi[0]);
        lo[1] = (__bf16)(f1[u] - (float)hi[1]);
        *(v2bf*)&Bhi[buf][n][kp * 2] = hi;   // ds_store_b32 (k, k+1 packed)
        *(v2bf*)&Blo[buf][n][kp * 2] = lo;
      }
    }
  };

  v8f c[4];
#pragma unroll
  for (int nt = 0; nt < 4; ++nt)
#pragma unroll
    for (int j = 0; j < 8; ++j) c[nt][j] = 0.f;

  const int KT = Din / BK;

  prefetch(0);
  stage(0);
  __syncthreads();

  for (int kt = 0; kt < KT; ++kt) {
    const int cur = kt & 1;
    if (kt + 1 < KT) prefetch((kt + 1) * BK);   // global loads overlap compute

    // ---- A fragments (ISA 16-bit A 16x32 layout)
    const int mrow  = (wid << 4) + l15;
    const int abase = up ? 8 : 0;
    Frag ahi, alo;
    ahi.h[0] = *(const v8bf*)&Ahi[cur][mrow][abase];
    ahi.h[1] = *(const v8bf*)&Ahi[cur][mrow][16 + abase];
    alo.h[0] = *(const v8bf*)&Alo[cur][mrow][abase];
    alo.h[1] = *(const v8bf*)&Alo[cur][mrow][16 + abase];

    const int kb = up ? 8 : 0;
#pragma unroll
    for (int nt = 0; nt < 4; ++nt) {
      const int n = (nt << 4) + l15;
      Frag bhi, blo;
      bhi.h[0] = *(const v8bf*)&Bhi[cur][n][kb];
      bhi.h[1] = *(const v8bf*)&Bhi[cur][n][kb + 16];
      blo.h[0] = *(const v8bf*)&Blo[cur][n][kb];
      blo.h[1] = *(const v8bf*)&Blo[cur][n][kb + 16];

      // bf16x3: x*y ~= hi*hi + hi*lo + lo*hi, f32 accumulate
      c[nt] = __builtin_amdgcn_wmma_f32_16x16x32_bf16(
                  false, ahi.v, false, bhi.v, (short)0, c[nt], false, false);
      c[nt] = __builtin_amdgcn_wmma_f32_16x16x32_bf16(
                  false, ahi.v, false, blo.v, (short)0, c[nt], false, false);
      c[nt] = __builtin_amdgcn_wmma_f32_16x16x32_bf16(
                  false, alo.v, false, bhi.v, (short)0, c[nt], false, false);
    }

    if (kt + 1 < KT) stage(cur ^ 1);            // write the other buffer
    __syncthreads();
  }

  // ---- epilogue: bias (+ GELU), store (C/D layout: lane<16 -> M=vgpr, else M=vgpr+8)
#pragma unroll
  for (int nt = 0; nt < 4; ++nt) {
    const int n = n0 + (nt << 4) + l15;
    const float b = Bias[e * Dout + n];
#pragma unroll
    for (int i = 0; i < 8; ++i) {
      int m = (wid << 4) + (up ? i + 8 : i);
      if (mBase + m < cntE) {
        float v = c[nt][i] + b;
        if (APPLY_GELU)
          v = 0.5f * v * (1.f + erff(v * 0.70710678118654752f));
        Out[(size_t)(offE + mBase + m) * Dout + n] = v;
      }
    }
  }
}

// ---------------------------------------------------------------------------
// 6) combine: out[t] = s0 * y[pos[t,0]] + s1 * y[pos[t,1]]
// ---------------------------------------------------------------------------
__global__ __launch_bounds__(128)
void combine_kernel(const float* __restrict__ y,
                    const int*   __restrict__ pos,
                    const float* __restrict__ scores,
                    float*       __restrict__ out)
{
  const int t  = blockIdx.x;
  const int p0 = pos[t * 2 + 0], p1 = pos[t * 2 + 1];
  const float s0 = scores[t * 2 + 0], s1 = scores[t * 2 + 1];
  const float4* y0 = (const float4*)(y + (size_t)p0 * D_MODEL);
  const float4* y1 = (const float4*)(y + (size_t)p1 * D_MODEL);
  float4* o = (float4*)(out + (size_t)t * D_MODEL);
  const int i = threadIdx.x;
  float4 a = y0[i], b = y1[i], r;
  r.x = s0 * a.x + s1 * b.x;
  r.y = s0 * a.y + s1 * b.y;
  r.z = s0 * a.z + s1 * b.z;
  r.w = s0 * a.w + s1 * b.w;
  o[i] = r;
}

// ---------------------------------------------------------------------------
extern "C" void kernel_launch(void* const* d_in, const int* in_sizes, int n_in,
                              void* d_out, int out_size, void* d_ws, size_t ws_size,
                              hipStream_t stream) {
  const float* x      = (const float*)d_in[0];
  const float* gate_w = (const float*)d_in[1];
  const float* gate_b = (const float*)d_in[2];
  const float* w1     = (const float*)d_in[3];
  const float* b1     = (const float*)d_in[4];
  const float* w2     = (const float*)d_in[5];
  const float* b2     = (const float*)d_in[6];
  float* out = (float*)d_out;

  // workspace layout (~192.4 MB): y aliases xs (xs dead after GEMM1)
  float* xs      = (float*)d_ws;                          // [N_ROWS, D_MODEL]
  float* h       = xs + (size_t)N_ROWS * D_MODEL;         // [N_ROWS, H_DIM]
  float* scores  = h  + (size_t)N_ROWS * H_DIM;           // [N_ROWS]
  int*   top_idx = (int*)(scores + N_ROWS);               // [N_ROWS]
  int*   pos     = top_idx + N_ROWS;                      // [N_ROWS]
  int*   cnt     = pos + N_ROWS;                          // [N_EXP]
  int*   offs    = cnt + N_EXP;                           // [N_EXP]
  int*   fill    = offs + N_EXP;                          // [N_EXP]
  float* y       = xs;                                    // alias

  init_kernel<<<1, 32, 0, stream>>>(cnt, fill);
  gate_kernel<<<T_TOK / 8, 256, 0, stream>>>(x, gate_w, gate_b, top_idx, scores, cnt);
  offsets_kernel<<<1, 1, 0, stream>>>(cnt, offs);
  scatter_kernel<<<N_ROWS, 128, 0, stream>>>(x, top_idx, offs, fill, pos, xs);

  dim3 g1(N_ROWS / BM, H_DIM / BN, N_EXP);     // m-tiles over-provisioned, early-exit
  moe_grouped_gemm<true ><<<g1, GTHREADS, 0, stream>>>(xs, w1, b1, h, offs, cnt, D_MODEL, H_DIM);
  dim3 g2(N_ROWS / BM, D_MODEL / BN, N_EXP);
  moe_grouped_gemm<false><<<g2, GTHREADS, 0, stream>>>(h,  w2, b2, y, offs, cnt, H_DIM, D_MODEL);

  combine_kernel<<<T_TOK, 128, 0, stream>>>(y, pos, scores, out);
}
